// TDNN_70703751626893
// MI455X (gfx1250) — matrix operand order repaired and checked by reading, not compiled, not run on hardware
//
#include <hip/hip_runtime.h>
#include <hip/hip_bf16.h>

// TDNN / 1D conv as 5 shifted GEMMs, bf16x3 split precision on
// v_wmma_f32_16x16x32_bf16 (CDNA5 / gfx1250, wave32).
//
//  * weights pre-split once into bf16 hi/lo, [k][o][i] layout (d_ws) ->
//    hot-loop W staging becomes coalesced 16B copies instead of stride-20B
//    f32 gathers + VALU converts repeated by every block
//  * double-buffered LDS pipeline (stage chunk ic+1 under WMMAs of chunk ic)
//  * W copies use GLOBAL_LOAD_ASYNC_TO_LDS_B128 (ASYNCcnt); signature
//    probe-confirmed: (v4i AS1*, v4i AS3*, imm offset, imm cpol)

#define B_   32
#define T_   2048
#define DIN  512
#define DOUT 512
#define KW   5
#define TOUT (T_ - KW + 1)       // 2044

#define TM    64                 // time tile per block
#define TN    64                 // d_out tile per block
#define KC    32                 // input-channel chunk (WMMA K)
#define XROWS (TM + KW - 1)      // 68 (halo for the 5 shifts)
#define PITCH 40                 // LDS pitch in bf16 (80B: 16B-aligned, spreads banks)
#define NCHUNK (DIN / KC)        // 16

typedef __attribute__((ext_vector_type(16))) __bf16       v16bf;
typedef __attribute__((ext_vector_type(8)))  float        v8f;
typedef __attribute__((ext_vector_type(4)))  unsigned int v4u;
typedef __attribute__((ext_vector_type(4)))  int          v4i;

// builtin parameter types per the round-2 diagnostic:
typedef __attribute__((address_space(1))) v4i gas_v4i;   // global
typedef __attribute__((address_space(3))) v4i las_v4i;   // LDS

#if defined(__has_builtin)
#  if __has_builtin(__builtin_amdgcn_global_load_async_to_lds_b128)
#    define USE_ASYNC_LDS 1
#  endif
#endif
#ifndef USE_ASYNC_LDS
#  define USE_ASYNC_LDS 0
#endif

union Frag { v16bf v; v4u q[2]; };

static __device__ __forceinline__ v8f wmma_bf16(const Frag& a, const Frag& b, v8f c) {
    return __builtin_amdgcn_wmma_f32_16x16x32_bf16(
        /*neg_a=*/false, a.v, /*neg_b=*/false, b.v,
        /*c_mod=*/(short)0, c, /*reuse_a=*/false, /*reuse_b=*/false);
}

// 16-byte global -> LDS copy; async (ASYNCcnt) when the toolchain exposes it.
static __device__ __forceinline__ void cp16_g2l(const __bf16* g, __bf16* l) {
#if USE_ASYNC_LDS
    __builtin_amdgcn_global_load_async_to_lds_b128(
        (gas_v4i*)g, (las_v4i*)l, /*offset=*/0, /*cpol=*/0);
#else
    *(v4u*)l = *(const v4u*)g;
#endif
}

static __device__ __forceinline__ void wait_async() {
#if USE_ASYNC_LDS
#  if __has_builtin(__builtin_amdgcn_s_wait_asynccnt)
    __builtin_amdgcn_s_wait_asynccnt(0);
#  else
    asm volatile("s_wait_asynccnt 0x0" ::: "memory");
#  endif
#endif
}

// ---- staging helpers -------------------------------------------------------

static __device__ __forceinline__ void stage_x(const float* __restrict__ x,
                                               int b, int t0, int ic, int tid,
                                               __bf16* xh, __bf16* xl) {
    for (int idx = tid; idx < XROWS * KC; idx += 256) {
        int r = idx >> 5, c = idx & 31;
        int t = t0 + r;
        float v = (t < T_) ? x[((size_t)b * T_ + t) * DIN + ic * KC + c] : 0.0f;
        __bf16 h = (__bf16)v;
        xh[r * PITCH + c] = h;
        xl[r * PITCH + c] = (__bf16)(v - (float)h);
    }
}

// pre-split path: whig/wlog are bf16 in [k][o][i] layout -> coalesced 16B copies
static __device__ __forceinline__ void stage_w_split(const __bf16* __restrict__ whig,
                                                     const __bf16* __restrict__ wlog,
                                                     int o0, int ic, int tid,
                                                     __bf16* wh, __bf16* wl) {
    for (int idx = tid; idx < KW * TN * 4; idx += 256) {   // 1280 16B chunks, exact x5
        int c = idx & 3, row = idx >> 2;
        int ol = row & 63, k = row >> 6;
        size_t goff = ((size_t)k * DOUT + o0 + ol) * DIN + ic * KC + c * 8;
        int    loff = (k * TN + ol) * PITCH + c * 8;
        cp16_g2l(&whig[goff], &wh[loff]);
        cp16_g2l(&wlog[goff], &wl[loff]);
    }
}

// fallback path (workspace too small): convert f32 [o][i][k] in-kernel
static __device__ __forceinline__ void stage_w_convert(const float* __restrict__ w,
                                                       int o0, int ic, int tid,
                                                       __bf16* wh, __bf16* wl) {
    for (int idx = tid; idx < KW * TN * KC; idx += 256) {
        int c = idx & 31, rest = idx >> 5;
        int ol = rest & 63, k = rest >> 6;
        float v = w[(((size_t)(o0 + ol)) * DIN + ic * KC + c) * KW + k];
        __bf16 h = (__bf16)v;
        wh[(k * TN + ol) * PITCH + c] = h;
        wl[(k * TN + ol) * PITCH + c] = (__bf16)(v - (float)h);
    }
}

// ---- one-time weight split: w[o][i][k] f32 -> whi/wlo[k][o][i] bf16 --------

__global__ __launch_bounds__(256)
void split_weights(const float* __restrict__ w,
                   __bf16* __restrict__ whi, __bf16* __restrict__ wlo) {
    int idx = blockIdx.x * 256 + threadIdx.x;      // over KW*DOUT*DIN, i fastest
    if (idx >= KW * DOUT * DIN) return;
    int i = idx & (DIN - 1);
    int rest = idx >> 9;
    int o = rest & (DOUT - 1);
    int k = rest >> 9;
    float v = w[((size_t)o * DIN + i) * KW + k];
    __bf16 h = (__bf16)v;
    whi[idx] = h;
    wlo[idx] = (__bf16)(v - (float)h);
}

// ---- main kernel -----------------------------------------------------------

__global__ __launch_bounds__(256)
void tdnn_conv1d_wmma_bf16x3(const float* __restrict__ x,
                             const float* __restrict__ w,
                             const float* __restrict__ bias,
                             const __bf16* __restrict__ whig,
                             const __bf16* __restrict__ wlog,
                             int presplit,
                             float* __restrict__ out)
{
    __shared__ __align__(16) __bf16 sXhi[2][XROWS * PITCH];
    __shared__ __align__(16) __bf16 sXlo[2][XROWS * PITCH];
    __shared__ __align__(16) __bf16 sWhi[2][KW * TN * PITCH];
    __shared__ __align__(16) __bf16 sWlo[2][KW * TN * PITCH];

    const int tid  = threadIdx.x;
    const int lane = tid & 31;
    const int wave = tid >> 5;
    const int mt   = wave & 3;          // which 16-row m strip of the 64
    const int ng   = wave >> 2;         // which pair of 16-col n tiles

    const int t0 = blockIdx.x * TM;
    const int o0 = blockIdx.y * TN;
    const int b  = blockIdx.z;

    const int half = lane >> 4;         // 0: lanes 0-15, 1: lanes 16-31
    const int lrow = lane & 15;

    // Per-lane fragment base offsets (ISA 16-bit A 16x32 / B 32x16 layouts):
    //   A lane: row M=lrow, K in {half*8 .. +7} and {half*8+16 .. +23}
    //   B lane: row N=lrow, K in {half*8 .. +7} and {half*8+16 .. +23}
    const int aBase = (mt * 16 + lrow) * PITCH + half * 8;
    const int bBase = lrow * PITCH + half * 8;

    v8f acc[2] = { v8f{}, v8f{} };

    // prologue: stage chunk 0 into buffer 0
    stage_x(x, b, t0, 0, tid, sXhi[0], sXlo[0]);
    if (presplit) stage_w_split(whig, wlog, o0, 0, tid, sWhi[0], sWlo[0]);
    else          stage_w_convert(w,       o0, 0, tid, sWhi[0], sWlo[0]);
    wait_async();
    __syncthreads();

    for (int ic = 0; ic < NCHUNK; ++ic) {
        const int cur = ic & 1;
        const int nxt = cur ^ 1;

        // stage next chunk under this chunk's WMMAs (other LDS buffer)
        if (ic + 1 < NCHUNK) {
            stage_x(x, b, t0, ic + 1, tid, sXhi[nxt], sXlo[nxt]);
            if (presplit) stage_w_split(whig, wlog, o0, ic + 1, tid, sWhi[nxt], sWlo[nxt]);
            else          stage_w_convert(w,       o0, ic + 1, tid, sWhi[nxt], sWlo[nxt]);
        }

        const __bf16* xh = sXhi[cur];
        const __bf16* xl = sXlo[cur];
        const __bf16* wh = sWhi[cur];
        const __bf16* wl = sWlo[cur];

#pragma unroll
        for (int k = 0; k < KW; ++k) {
            Frag ahi, alo;
            const int aoff = k * PITCH + aBase;       // row = k + mt*16 + lrow
            ahi.q[0] = *(const v4u*)&xh[aoff];
            ahi.q[1] = *(const v4u*)&xh[aoff + 16];
            alo.q[0] = *(const v4u*)&xl[aoff];
            alo.q[1] = *(const v4u*)&xl[aoff + 16];
#pragma unroll
            for (int j = 0; j < 2; ++j) {
                const int nt   = ng * 2 + j;
                const int boff = (k * TN + nt * 16) * PITCH + bBase;
                Frag bhi, blo;
                bhi.q[0] = *(const v4u*)&wh[boff];
                bhi.q[1] = *(const v4u*)&wh[boff + 16];
                blo.q[0] = *(const v4u*)&wl[boff];
                blo.q[1] = *(const v4u*)&wl[boff + 16];
                acc[j] = wmma_bf16(ahi, bhi, acc[j]);   // hi*hi
                acc[j] = wmma_bf16(ahi, blo, acc[j]);   // hi*lo
                acc[j] = wmma_bf16(alo, bhi, acc[j]);   // lo*hi
            }
        }

        wait_async();          // our async W copies into [nxt] have landed
        __syncthreads();       // publish [nxt] to all waves; retire [cur]
    }

    // ---- epilogue: bias + relu, store (out is [b][o][t], t contiguous) ----
#pragma unroll
    for (int j = 0; j < 2; ++j) {
        const int nt = ng * 2 + j;
        const int o  = o0 + nt * 16 + lrow;
        const float bo = bias[o];
        const int tb = t0 + mt * 16 + half * 8;        // C layout: lanes>=16 hold M=r+8
        float* op = &out[((size_t)b * DOUT + o) * TOUT + tb];
#pragma unroll
        for (int r = 0; r < 8; ++r) {
            int t = tb + r;
            if (t < TOUT) {
                float v = acc[j][r] + bo;
                op[r] = v > 0.0f ? v : 0.0f;
            }
        }
    }
}

// ---- host side -------------------------------------------------------------

extern "C" void kernel_launch(void* const* d_in, const int* in_sizes, int n_in,
                              void* d_out, int out_size, void* d_ws, size_t ws_size,
                              hipStream_t stream) {
    (void)in_sizes; (void)n_in; (void)out_size;
    const float* x    = (const float*)d_in[0];
    const float* w    = (const float*)d_in[1];
    const float* bias = (const float*)d_in[2];
    float* out        = (float*)d_out;

    const size_t welems = (size_t)KW * DOUT * DIN;         // 1,310,720
    const int presplit = (d_ws != nullptr && ws_size >= welems * 2 * 2) ? 1 : 0;
    __bf16* whi = (__bf16*)d_ws;
    __bf16* wlo = presplit ? (whi + welems) : nullptr;

    if (presplit) {
        int n = (int)welems;
        split_weights<<<(n + 255) / 256, 256, 0, stream>>>(w, whi, wlo);
    }

    dim3 grid((TOUT + TM - 1) / TM,   // 32 time tiles
              DOUT / TN,              // 8 d_out tiles
              B_);                    // 32 batches
    dim3 block(256);                  // 8 wave32s
    tdnn_conv1d_wmma_bf16x3<<<grid, block, 0, stream>>>(
        x, w, bias, presplit ? whi : nullptr, presplit ? wlo : nullptr, presplit, out);
}
